// Model_62397284876556
// MI455X (gfx1250) — compile-verified, matrix operand order checked by reading, not run on hardware
//
#include <hip/hip_runtime.h>
#include <hip/hip_bf16.h>
#include <math.h>

// ---------------- model dims ----------------
#define B_      4
#define L_      1024
#define CIN     32
#define DMODEL  512
#define DINNER  1024
#define DSTATE  16
#define DTRANK  32
#define NCLS    10
#define BL      (B_ * L_)          // 4096 rows through every GEMM
#define NCHUNK  32                 // scan chunks
#define CLEN    32                 // steps per chunk (NCHUNK*CLEN == L_)

typedef __attribute__((ext_vector_type(16))) __bf16          v16bf;
typedef __attribute__((ext_vector_type(8)))  float           v8f;
typedef __attribute__((ext_vector_type(8)))  unsigned short  us8;

union Frag { us8 u[2]; v16bf v; };

__device__ __forceinline__ unsigned short f2bf(float f) {
    unsigned int u = __float_as_uint(f);
    unsigned int r = u + 0x7fffu + ((u >> 16) & 1u);   // RNE
    return (unsigned short)(r >> 16);
}
__device__ __forceinline__ float sigmoidf(float x) { return 1.0f / (1.0f + __expf(-x)); }

// ---------------- f32 -> bf16 (strided rows) ----------------
__global__ void cvt_bf16_kernel(const float* __restrict__ src,
                                unsigned short* __restrict__ dst,
                                int rows, int cols, int stride) {
    long long idx = (long long)blockIdx.x * blockDim.x + threadIdx.x;
    if (idx >= (long long)rows * cols) return;
    int r = (int)(idx / cols), c = (int)(idx % cols);
    dst[idx] = f2bf(src[(size_t)r * stride + c]);
}

// ---------------- input conv (edge pad, k=3) + positional emb ----------------
__global__ void embed_kernel(const float* __restrict__ x_enc,
                             const float* __restrict__ conv_w,
                             unsigned short* __restrict__ x_bf) {
    const int b = blockIdx.x >> 10;
    const int t = blockIdx.x & 1023;
    __shared__ float s_in[3 * CIN];
    if (threadIdx.x < 3 * CIN) {
        int j = threadIdx.x / CIN, c = threadIdx.x % CIN;
        int tt = t + j - 2; tt = tt < 0 ? 0 : (tt > L_ - 1 ? L_ - 1 : tt);  // edge pad
        s_in[threadIdx.x] = x_enc[((size_t)b * L_ + tt) * CIN + c];
    }
    __syncthreads();
    const float ln1e4 = 9.2103403719761836f;  // log(10000)
    unsigned short* out = x_bf + ((size_t)b * L_ + t) * DMODEL;
    #pragma unroll
    for (int i = 0; i < 4; ++i) {
        int dd = threadIdx.x * 4 + i;
        const float* w = conv_w + (size_t)dd * (CIN * 3);
        float acc = 0.f;
        #pragma unroll 8
        for (int c = 0; c < CIN; ++c) {
            acc += w[c * 3 + 0] * s_in[0 * CIN + c];
            acc += w[c * 3 + 1] * s_in[1 * CIN + c];
            acc += w[c * 3 + 2] * s_in[2 * CIN + c];
        }
        float freq = __expf(-ln1e4 * (float)(dd & ~1) / (float)DMODEL);
        float ang  = (float)t * freq;
        acc += (dd & 1) ? __cosf(ang) : __sinf(ang);
        out[dd] = f2bf(acc);
    }
}

// ---------------- WMMA GEMM: C[M,N] = A[M,K] * W[N,K]^T, bf16 in / f32 out ----
// 256 threads = 8 waves; each wave owns a 32x32 output block (2x2 WMMA tiles),
// block covers 128(M) x 64(N); grid = (N/64, M/128).
__global__ void wmma_gemm_bf16(const unsigned short* __restrict__ A,
                               const unsigned short* __restrict__ W,
                               float* __restrict__ C,
                               int M, int N, int K) {
    const int lane  = threadIdx.x & 31;
    const int wave  = threadIdx.x >> 5;                  // 0..7
    const int mBase = blockIdx.y * 128 + (wave >> 1) * 32;
    const int nBase = blockIdx.x * 64  + (wave & 1) * 32;
    const int half  = lane >> 4;          // K-half selector per ISA 16-bit A layout
    const int l15   = lane & 15;
    const unsigned short* pa0 = A + (size_t)(mBase + l15) * K + half * 8;
    const unsigned short* pa1 = pa0 + (size_t)16 * K;
    const unsigned short* pb0 = W + (size_t)(nBase + l15) * K + half * 8;
    const unsigned short* pb1 = pb0 + (size_t)16 * K;
    v8f acc00 = {}, acc01 = {}, acc10 = {}, acc11 = {};
    for (int kk = 0; kk < K; kk += 32) {
        Frag a0, a1, b0, b1;
        a0.u[0] = *(const us8*)(pa0 + kk);  a0.u[1] = *(const us8*)(pa0 + kk + 16);
        a1.u[0] = *(const us8*)(pa1 + kk);  a1.u[1] = *(const us8*)(pa1 + kk + 16);
        b0.u[0] = *(const us8*)(pb0 + kk);  b0.u[1] = *(const us8*)(pb0 + kk + 16);
        b1.u[0] = *(const us8*)(pb1 + kk);  b1.u[1] = *(const us8*)(pb1 + kk + 16);
        acc00 = __builtin_amdgcn_wmma_f32_16x16x32_bf16(false, a0.v, false, b0.v, (short)0, acc00, false, false);
        acc01 = __builtin_amdgcn_wmma_f32_16x16x32_bf16(false, a0.v, false, b1.v, (short)0, acc01, false, false);
        acc10 = __builtin_amdgcn_wmma_f32_16x16x32_bf16(false, a1.v, false, b0.v, (short)0, acc10, false, false);
        acc11 = __builtin_amdgcn_wmma_f32_16x16x32_bf16(false, a1.v, false, b1.v, (short)0, acc11, false, false);
    }
    // C/D layout: elem r -> row mb + r + (half?8:0), col nb + l15
    #pragma unroll
    for (int tile = 0; tile < 4; ++tile) {
        const v8f acc = (tile == 0) ? acc00 : (tile == 1) ? acc01 : (tile == 2) ? acc10 : acc11;
        const int mb = mBase + (tile >> 1) * 16;
        const int nb = nBase + (tile & 1) * 16;
        float* pc = C + (size_t)(mb + half * 8) * N + nb + l15;
        #pragma unroll
        for (int r = 0; r < 8; ++r) pc[(size_t)r * N] = acc[r];
    }
}

// ---------------- depthwise causal conv (D_CONV=4) + bias + SiLU ----------------
__global__ void dwconv_silu_kernel(const float* __restrict__ xz,
                                   const float* __restrict__ w,
                                   const float* __restrict__ bias,
                                   float* __restrict__ xi_f,
                                   unsigned short* __restrict__ xi_bf) {
    int idx = blockIdx.x * blockDim.x + threadIdx.x;     // < BL*DINNER
    int ch = idx & (DINNER - 1);
    int r  = idx >> 10;                                  // b*L + t
    int t  = r & (L_ - 1);
    int b  = r >> 10;
    const float* wc = w + (size_t)ch * 4;
    float acc = bias[ch];
    #pragma unroll
    for (int j = 0; j < 4; ++j) {
        int tj = t - 3 + j;
        if (tj >= 0)
            acc += wc[j] * xz[((size_t)b * L_ + tj) * (2 * DINNER) + ch];
    }
    float s = acc * sigmoidf(acc);
    xi_f[idx]  = s;
    xi_bf[idx] = f2bf(s);
}

// ---------------- softplus(x + bias) epilogue for dt_proj ----------------
__global__ void softplus_kernel(float* __restrict__ delta,
                                const float* __restrict__ bias) {
    int idx = blockIdx.x * blockDim.x + threadIdx.x;     // < BL*DINNER
    int ch = idx & (DINNER - 1);
    float x = delta[idx] + bias[ch];
    delta[idx] = (x > 20.f) ? x : log1pf(__expf(x));
}

// ================= chunked parallel selective scan =================
// recurrence h_t = exp(d_t*A)*h_{t-1} + d_t*x_t*B_t is linear in h, so a
// chunk's transfer is h_out = exp(A*sum(d)) * h_in + r, with r the chunk run
// from h=0.  Phase1 (131072 threads): per-chunk (r, sum_d).  Phase2 (4096
// threads): 32-step combine, storing each chunk's entry state.  Phase3
// (131072 threads): replay chunk from entry state, emit gated bf16 output.

// thread -> (b, chunk, ch):  ch = tid&1023, chunk = (tid>>10)&31, b = tid>>15
__global__ void scan_chunk_kernel(const float* __restrict__ delta,
                                  const float* __restrict__ xi,
                                  const float* __restrict__ dbc,
                                  const float* __restrict__ A_log,
                                  float* __restrict__ chunk_r,
                                  float* __restrict__ chunk_sd) {
    int tid   = blockIdx.x * blockDim.x + threadIdx.x;
    int ch    = tid & (DINNER - 1);
    int rest  = tid >> 10;
    int chunk = rest & (NCHUNK - 1);
    int b     = rest >> 5;
    float A[DSTATE], h[DSTATE];
    #pragma unroll
    for (int s = 0; s < DSTATE; ++s) { A[s] = -__expf(A_log[ch * DSTATE + s]); h[s] = 0.f; }
    const int t0 = chunk * CLEN;
    const float* dl = delta + ((size_t)b * L_ + t0) * DINNER + ch;
    const float* xl = xi    + ((size_t)b * L_ + t0) * DINNER + ch;
    const float* bc = dbc   + ((size_t)b * L_ + t0) * (DTRANK + 2 * DSTATE);
    float Sd = 0.f;
    for (int t = 0; t < CLEN; ++t) {
        float d  = dl[(size_t)t * DINNER];
        float xv = xl[(size_t)t * DINNER];
        const float* r = bc + (size_t)t * (DTRANK + 2 * DSTATE);
        float dx = d * xv;
        Sd += d;
        #pragma unroll
        for (int s = 0; s < DSTATE; ++s)
            h[s] = __expf(d * A[s]) * h[s] + dx * r[DTRANK + s];
    }
    size_t base = ((size_t)(b * NCHUNK + chunk) * DSTATE) * DINNER + ch;
    #pragma unroll
    for (int s = 0; s < DSTATE; ++s) chunk_r[base + (size_t)s * DINNER] = h[s];
    chunk_sd[(size_t)(b * NCHUNK + chunk) * DINNER + ch] = Sd;
}

// thread -> (b, ch): sequential over the 32 chunk summaries
__global__ void scan_combine_kernel(const float* __restrict__ chunk_r,
                                    const float* __restrict__ chunk_sd,
                                    const float* __restrict__ A_log,
                                    float* __restrict__ chunk_h0) {
    int tid = blockIdx.x * blockDim.x + threadIdx.x;   // < B_*DINNER
    int ch = tid & (DINNER - 1), b = tid >> 10;
    float A[DSTATE], h[DSTATE];
    #pragma unroll
    for (int s = 0; s < DSTATE; ++s) { A[s] = -__expf(A_log[ch * DSTATE + s]); h[s] = 0.f; }
    for (int c = 0; c < NCHUNK; ++c) {
        size_t base = ((size_t)(b * NCHUNK + c) * DSTATE) * DINNER + ch;
        #pragma unroll
        for (int s = 0; s < DSTATE; ++s) chunk_h0[base + (size_t)s * DINNER] = h[s];
        float Sd = chunk_sd[(size_t)(b * NCHUNK + c) * DINNER + ch];
        #pragma unroll
        for (int s = 0; s < DSTATE; ++s)
            h[s] = __expf(A[s] * Sd) * h[s] + chunk_r[base + (size_t)s * DINNER];
    }
}

// thread -> (b, chunk, ch): replay chunk from entry state, fuse gating epilogue
__global__ void scan_emit_kernel(const float* __restrict__ delta,
                                 const float* __restrict__ xi,
                                 const float* __restrict__ dbc,
                                 const float* __restrict__ xz,
                                 const float* __restrict__ A_log,
                                 const float* __restrict__ Dp,
                                 const float* __restrict__ chunk_h0,
                                 unsigned short* __restrict__ y_bf) {
    int tid   = blockIdx.x * blockDim.x + threadIdx.x;
    int ch    = tid & (DINNER - 1);
    int rest  = tid >> 10;
    int chunk = rest & (NCHUNK - 1);
    int b     = rest >> 5;
    float A[DSTATE], h[DSTATE];
    size_t base = ((size_t)(b * NCHUNK + chunk) * DSTATE) * DINNER + ch;
    #pragma unroll
    for (int s = 0; s < DSTATE; ++s) {
        A[s] = -__expf(A_log[ch * DSTATE + s]);
        h[s] = chunk_h0[base + (size_t)s * DINNER];
    }
    const float dp = Dp[ch];
    const int t0 = chunk * CLEN;
    const float* dl = delta + ((size_t)b * L_ + t0) * DINNER + ch;
    const float* xl = xi    + ((size_t)b * L_ + t0) * DINNER + ch;
    const float* zl = xz    + ((size_t)b * L_ + t0) * (2 * DINNER) + DINNER + ch;
    const float* bc = dbc   + ((size_t)b * L_ + t0) * (DTRANK + 2 * DSTATE);
    unsigned short* yl = y_bf + ((size_t)b * L_ + t0) * DINNER + ch;
    for (int t = 0; t < CLEN; ++t) {
        float d  = dl[(size_t)t * DINNER];
        float xv = xl[(size_t)t * DINNER];
        const float* r = bc + (size_t)t * (DTRANK + 2 * DSTATE);
        float dx = d * xv, y = 0.f;
        #pragma unroll
        for (int s = 0; s < DSTATE; ++s) {
            h[s] = __expf(d * A[s]) * h[s] + dx * r[DTRANK + s];        // B
            y   += h[s] * r[DTRANK + DSTATE + s];                       // C
        }
        float zv = zl[(size_t)t * (2 * DINNER)];
        float out = (y + xv * dp) * (zv * sigmoidf(zv));
        yl[(size_t)t * DINNER] = f2bf(out);
    }
}

// ---------------- LayerNorm + class head + importance head ----------------
__device__ __forceinline__ float red128_sum(float v, float* red) {
    red[threadIdx.x] = v; __syncthreads();
    for (int o = 64; o > 0; o >>= 1) {
        if (threadIdx.x < o) red[threadIdx.x] += red[threadIdx.x + o];
        __syncthreads();
    }
    float r = red[0]; __syncthreads();
    return r;
}

__global__ void ln_heads_kernel(const float* __restrict__ mo,
                                const float* __restrict__ ln_w, const float* __restrict__ ln_b,
                                const float* __restrict__ ol_w,
                                const float* __restrict__ if_w, const float* __restrict__ if_b,
                                float* __restrict__ logits, float* __restrict__ imp) {
    __shared__ float srow[DMODEL];
    __shared__ float red[128];
    const int row = blockIdx.x;
    const float* pr = mo + (size_t)row * DMODEL;
    float v[4], s = 0.f, s2 = 0.f;
    #pragma unroll
    for (int i = 0; i < 4; ++i) { v[i] = pr[threadIdx.x * 4 + i]; s += v[i]; s2 += v[i] * v[i]; }
    float mean = red128_sum(s,  red) * (1.0f / DMODEL);
    float ex2  = red128_sum(s2, red) * (1.0f / DMODEL);
    float rinv = rsqrtf(ex2 - mean * mean + 1e-5f);
    #pragma unroll
    for (int i = 0; i < 4; ++i) {
        int d = threadIdx.x * 4 + i;
        srow[d] = (v[i] - mean) * rinv * ln_w[d] + ln_b[d];
    }
    __syncthreads();
    for (int c = 0; c < NCLS; ++c) {
        const float* w = ol_w + (size_t)c * DMODEL;
        float p = 0.f;
        #pragma unroll
        for (int i = 0; i < 4; ++i) { int d = threadIdx.x * 4 + i; p += srow[d] * w[d]; }
        float tot = red128_sum(p, red);
        if (threadIdx.x == 0) logits[(size_t)row * NCLS + c] = tot;
    }
    float p = 0.f;
    #pragma unroll
    for (int i = 0; i < 4; ++i) { int d = threadIdx.x * 4 + i; p += srow[d] * if_w[d]; }
    float tot = red128_sum(p, red);
    if (threadIdx.x == 0) imp[row] = tot + if_b[0];
}

// ---------------- masked softmax over L + weighted class reduction ----------------
__global__ void final_kernel(const float* __restrict__ logits,
                             const float* __restrict__ imp,
                             const float* __restrict__ mark,
                             float* __restrict__ out) {
    __shared__ float red[256];
    const int b = blockIdx.x;
    const float* ip = imp  + (size_t)b * L_;
    const float* mk = mark + (size_t)b * L_;
    float m = -INFINITY;
    for (int t = threadIdx.x; t < L_; t += 256) m = fmaxf(m, ip[t]);
    red[threadIdx.x] = m; __syncthreads();
    for (int o = 128; o > 0; o >>= 1) {
        if (threadIdx.x < o) red[threadIdx.x] = fmaxf(red[threadIdx.x], red[threadIdx.x + o]);
        __syncthreads();
    }
    m = red[0]; __syncthreads();
    // out_c = sum_t e_t*mk_t^2*logits_tc / sum_t e_t*mk_t
    float E = 0.f, num[NCLS];
    #pragma unroll
    for (int c = 0; c < NCLS; ++c) num[c] = 0.f;
    for (int t = threadIdx.x; t < L_; t += 256) {
        float e  = __expf(ip[t] - m);
        float mv = mk[t];
        E += e * mv;
        float em2 = e * mv * mv;
        const float* lg = logits + ((size_t)b * L_ + t) * NCLS;
        #pragma unroll
        for (int c = 0; c < NCLS; ++c) num[c] += em2 * lg[c];
    }
    red[threadIdx.x] = E; __syncthreads();
    for (int o = 128; o > 0; o >>= 1) {
        if (threadIdx.x < o) red[threadIdx.x] += red[threadIdx.x + o];
        __syncthreads();
    }
    E = red[0]; __syncthreads();
    for (int c = 0; c < NCLS; ++c) {
        red[threadIdx.x] = num[c]; __syncthreads();
        for (int o = 128; o > 0; o >>= 1) {
            if (threadIdx.x < o) red[threadIdx.x] += red[threadIdx.x + o];
            __syncthreads();
        }
        if (threadIdx.x == 0) out[(size_t)b * NCLS + c] = red[0] / E;
        __syncthreads();
    }
}

// ================================================================
extern "C" void kernel_launch(void* const* d_in, const int* in_sizes, int n_in,
                              void* d_out, int out_size, void* d_ws, size_t ws_size,
                              hipStream_t stream) {
    const float* x_enc      = (const float*)d_in[0];
    const float* x_mark_enc = (const float*)d_in[1];
    const float* conv_w     = (const float*)d_in[4];
    const float* in_proj_w  = (const float*)d_in[5];
    const float* conv1d_w   = (const float*)d_in[6];
    const float* conv1d_b   = (const float*)d_in[7];
    const float* x_proj_w   = (const float*)d_in[8];
    const float* dt_proj_w  = (const float*)d_in[9];
    const float* dt_proj_b  = (const float*)d_in[10];
    const float* A_log      = (const float*)d_in[11];
    const float* Dp         = (const float*)d_in[12];
    const float* out_proj_w = (const float*)d_in[13];
    const float* ln_w       = (const float*)d_in[14];
    const float* ln_b       = (const float*)d_in[15];
    const float* out_lay_w  = (const float*)d_in[16];
    const float* if_w       = (const float*)d_in[17];
    const float* if_b       = (const float*)d_in[18];
    float* out = (float*)d_out;

    // ---- workspace layout (256B aligned slabs) ----
    char* p = (char*)d_ws;
    size_t off = 0;
    auto alloc = [&](size_t bytes) { void* q = p + off; off = (off + bytes + 255) & ~(size_t)255; return q; };
    unsigned short* x_bf   = (unsigned short*)alloc((size_t)BL * DMODEL * 2);
    unsigned short* w_in   = (unsigned short*)alloc((size_t)2 * DINNER * DMODEL * 2);
    unsigned short* w_xp   = (unsigned short*)alloc((size_t)(DTRANK + 2 * DSTATE) * DINNER * 2);
    unsigned short* w_dt   = (unsigned short*)alloc((size_t)DINNER * DTRANK * 2);
    unsigned short* w_out  = (unsigned short*)alloc((size_t)DMODEL * DINNER * 2);
    float*          xz     = (float*)alloc((size_t)BL * 2 * DINNER * 4);
    float*          xi_f   = (float*)alloc((size_t)BL * DINNER * 4);
    unsigned short* xi_bf  = (unsigned short*)alloc((size_t)BL * DINNER * 2);
    float*          dbc    = (float*)alloc((size_t)BL * (DTRANK + 2 * DSTATE) * 4);
    unsigned short* dt_bf  = (unsigned short*)alloc((size_t)BL * DTRANK * 2);
    float*          delta  = (float*)alloc((size_t)BL * DINNER * 4);
    float*          c_r    = (float*)alloc((size_t)B_ * NCHUNK * DSTATE * DINNER * 4);
    float*          c_sd   = (float*)alloc((size_t)B_ * NCHUNK * DINNER * 4);
    float*          c_h0   = (float*)alloc((size_t)B_ * NCHUNK * DSTATE * DINNER * 4);
    unsigned short* y_bf   = (unsigned short*)alloc((size_t)BL * DINNER * 2);
    float*          mo     = (float*)alloc((size_t)BL * DMODEL * 4);
    float*          logits = (float*)alloc((size_t)BL * NCLS * 4);
    float*          imp    = (float*)alloc((size_t)BL * 4);
    (void)ws_size; (void)in_sizes; (void)n_in; (void)out_size;

    auto cvt = [&](const float* src, unsigned short* dst, int rows, int cols, int stride) {
        long long n = (long long)rows * cols;
        cvt_bf16_kernel<<<dim3((unsigned)((n + 255) / 256)), dim3(256), 0, stream>>>(src, dst, rows, cols, stride);
    };
    // 1) weights -> bf16
    cvt(in_proj_w,  w_in,  2 * DINNER, DMODEL, DMODEL);
    cvt(x_proj_w,   w_xp,  DTRANK + 2 * DSTATE, DINNER, DINNER);
    cvt(dt_proj_w,  w_dt,  DINNER, DTRANK, DTRANK);
    cvt(out_proj_w, w_out, DMODEL, DINNER, DINNER);

    // 2) input conv + pos-emb -> x_bf
    embed_kernel<<<dim3(BL), dim3(128), 0, stream>>>(x_enc, conv_w, x_bf);

    // 3) in_proj GEMM: xz[4096,2048]
    wmma_gemm_bf16<<<dim3((2 * DINNER) / 64, BL / 128), dim3(256), 0, stream>>>(
        x_bf, w_in, xz, BL, 2 * DINNER, DMODEL);

    // 4) depthwise causal conv + SiLU
    dwconv_silu_kernel<<<dim3((BL * DINNER) / 256), dim3(256), 0, stream>>>(
        xz, conv1d_w, conv1d_b, xi_f, xi_bf);

    // 5) x_proj GEMM: dbc[4096,64]
    wmma_gemm_bf16<<<dim3((DTRANK + 2 * DSTATE) / 64, BL / 128), dim3(256), 0, stream>>>(
        xi_bf, w_xp, dbc, BL, DTRANK + 2 * DSTATE, DINNER);

    // 6) dt slice -> bf16 (row stride 64 in dbc)
    cvt(dbc, dt_bf, BL, DTRANK, DTRANK + 2 * DSTATE);

    // 7) dt_proj GEMM: delta[4096,1024]
    wmma_gemm_bf16<<<dim3(DINNER / 64, BL / 128), dim3(256), 0, stream>>>(
        dt_bf, w_dt, delta, BL, DINNER, DTRANK);

    // 8) delta = softplus(delta + dt_proj_b)
    softplus_kernel<<<dim3((BL * DINNER) / 256), dim3(256), 0, stream>>>(delta, dt_proj_b);

    // 9) chunked parallel scan (3 phases) -> y_bf
    scan_chunk_kernel<<<dim3((B_ * NCHUNK * DINNER) / 256), dim3(256), 0, stream>>>(
        delta, xi_f, dbc, A_log, c_r, c_sd);
    scan_combine_kernel<<<dim3((B_ * DINNER) / 256), dim3(256), 0, stream>>>(
        c_r, c_sd, A_log, c_h0);
    scan_emit_kernel<<<dim3((B_ * NCHUNK * DINNER) / 256), dim3(256), 0, stream>>>(
        delta, xi_f, dbc, xz, A_log, Dp, c_h0, y_bf);

    // 10) out_proj GEMM: mo[4096,512]
    wmma_gemm_bf16<<<dim3(DMODEL / 64, BL / 128), dim3(256), 0, stream>>>(
        y_bf, w_out, mo, BL, DMODEL, DINNER);

    // 11) LayerNorm + heads
    ln_heads_kernel<<<dim3(BL), dim3(128), 0, stream>>>(
        mo, ln_w, ln_b, out_lay_w, if_w, if_b, logits, imp);

    // 12) masked softmax weighting -> (4,10)
    final_kernel<<<dim3(B_), dim3(256), 0, stream>>>(logits, imp, x_mark_enc, out);
}
